// Downsample_66743791780383
// MI455X (gfx1250) — matrix operand order repaired and checked by reading, not compiled, not run on hardware
//
#include <hip/hip_runtime.h>
#include <hip/hip_bf16.h>

// ---------------------------------------------------------------------------
// Downsample: ternary-quantized 3x3 conv, stride 2, pad 1  (implicit GEMM)
//   x:(32,384,64,64) f32,  w:(384,384,3,3) f32 -> {-1,0,1},  bias:(384)
//   out:(32,384,32,32) f32
// GEMM: M=Cout, N=B*OH*OW, K=Cin*9 (ordered tap-major, channel-minor)
// Uses v_wmma_f32_16x16x32_f16 on gfx1250 (wave32).
// ---------------------------------------------------------------------------

typedef __attribute__((ext_vector_type(16))) _Float16 v16h;
typedef __attribute__((ext_vector_type(8)))  _Float16 v8h;
typedef __attribute__((ext_vector_type(8)))  float    v8f;

#define TERN_TH 0.001f

constexpr int Cin   = 384, Cout = 384;
constexpr int Hin   = 64,  Win  = 64;
constexpr int Hout  = 32,  Wout = 32;
constexpr int Batch = 32;

constexpr int CCHUNK = 32;                    // K-step channels per stage
constexpr int MTILE  = 32;                    // output channels per block
constexpr int NTILE  = 128;                   // output pixels per block (4 rows x 32)
constexpr int XROWS  = 9;                     // input halo rows  (2*4 + 1)
constexpr int XCOLS  = 66;                    // input halo cols  (64 + 2 pad)
constexpr int XSTRIDE = 32;                   // halves per (row,col) position
constexpr int WSTRIDE = 296;                  // halves per o-row (9*32 + 8 pad; 592B = 37*16)
constexpr int SX_HALVES = XROWS * XCOLS * XSTRIDE;   // 19008 -> 38016 B
constexpr int SW_HALVES = MTILE * WSTRIDE;           //  9472 -> 18944 B  (total 56960 B)

__global__ __launch_bounds__(256)
void downsample_ternary_wmma(const float* __restrict__ x,
                             const float* __restrict__ w,
                             const float* __restrict__ bias,
                             float* __restrict__ out)
{
    __shared__ _Float16 sX[SX_HALVES];   // [ (r*66+col) ][ cc ]  channel-innermost
    __shared__ _Float16 sW[SW_HALVES];   // [ o ][ p*32 + cc ]    padded row stride

    const int tid   = threadIdx.x;
    const int lane  = tid & 31;
    const int wv    = tid >> 5;                       // wave 0..7
    const int o0    = blockIdx.x * MTILE;             // 12 o-tiles (innermost -> L2 reuse of x)
    const int nb    = blockIdx.y * NTILE;             // 256 n-tiles
    const int b     = nb >> 10;                       // batch (128 | 1024)
    const int oh0   = (nb >> 5) & 31;                 // first output row of tile
    const int ih0   = 2 * oh0 - 1;                    // first input row of halo

    // per-lane output column (WMMA B/C mapping: N = lane % 16)
    const int nl    = lane & 15;
    const int n     = nb + wv * 16 + nl;
    const int oh    = (n >> 5) & 31;
    const int ow    = n & 31;
    const int rbase = 2 * (oh - oh0);                 // + kh -> halo row
    const int colb  = 2 * ow;                         // + kw -> halo col (col 0 == iw=-1)
    const int khalf = (lane >> 4) & 1;                // which K-half this lane carries

    v8f acc0 = {};
    v8f acc1 = {};

    const float* xb = x + (size_t)b * Cin * Hin * Win;

    for (int c0 = 0; c0 < Cin; c0 += CCHUNK) {
        __syncthreads();

        // ---- stage x halo chunk: global (coalesced over iw) -> LDS channel-innermost
        for (int i = tid; i < CCHUNK * XROWS * XCOLS; i += 256) {
            int cc  = i / (XROWS * XCOLS);
            int rem = i - cc * (XROWS * XCOLS);
            int r   = rem / XCOLS;
            int col = rem - r * XCOLS;
            int ih  = ih0 + r;
            int iw  = col - 1;
            float v = 0.0f;
            if ((unsigned)ih < (unsigned)Hin && (unsigned)iw < (unsigned)Win)
                v = xb[((size_t)(c0 + cc) * Hin + ih) * Win + iw];
            sX[(r * XCOLS + col) * XSTRIDE + cc] = (_Float16)v;
        }

        // ---- stage ternary-quantized weights for this (o-tile, c-chunk)
        for (int i = tid; i < MTILE * 9 * CCHUNK; i += 256) {
            int o   = i / (9 * CCHUNK);
            int rem = i - o * (9 * CCHUNK);
            int p   = rem >> 5;
            int cc  = rem & 31;
            float g = w[(size_t)(o0 + o) * (Cin * 9) + (size_t)(c0 + cc) * 9 + p];
            float q = (g > TERN_TH) ? 1.0f : ((g < -TERN_TH) ? -1.0f : 0.0f);
            sW[o * WSTRIDE + p * 32 + cc] = (_Float16)q;
        }

        // ---- hint next chunk of x into cache (gfx1250 global_prefetch_b8)
        if (c0 + CCHUNK < Cin) {
            const float* pf = &xb[((size_t)(c0 + CCHUNK + (tid & 31)) * Hin
                                   + (ih0 >= 0 ? ih0 : 0)) * Win + ((tid >> 5) << 3)];
            __builtin_prefetch(pf, 0, 1);
        }

        __syncthreads();

        // ---- 9 taps x (16x16x32 f16 WMMA) x 2 M-subtiles
        #pragma unroll
        for (int p = 0; p < 9; ++p) {
            const int kh = p / 3;
            const int kw = p - 3 * kh;

            // B fragment: lane carries N=nl, K = khalf*16 + e (contiguous channels)
            const int pos = (rbase + kh) * XCOLS + (colb + kw);
            const _Float16* bp = &sX[pos * XSTRIDE + khalf * 16];
            v8h b_lo = *(const v8h*)(bp);
            v8h b_hi = *(const v8h*)(bp + 8);
            v16h bm;
            #pragma unroll
            for (int e = 0; e < 8; ++e) { bm[e] = b_lo[e]; bm[8 + e] = b_hi[e]; }

            // A fragments: lane carries M=nl; K = khalf*16 + e (contiguous channels)
            const _Float16* ap0 = &sW[nl * WSTRIDE + p * 32 + khalf * 16];
            const _Float16* ap1 = &sW[(16 + nl) * WSTRIDE + p * 32 + khalf * 16];
            v8h a0lo = *(const v8h*)(ap0);
            v8h a0hi = *(const v8h*)(ap0 + 8);
            v8h a1lo = *(const v8h*)(ap1);
            v8h a1hi = *(const v8h*)(ap1 + 8);
            v16h am0, am1;
            #pragma unroll
            for (int e = 0; e < 8; ++e) {
                am0[e] = a0lo[e]; am0[8 + e] = a0hi[e];
                am1[e] = a1lo[e]; am1[8 + e] = a1hi[e];
            }

            acc0 = __builtin_amdgcn_wmma_f32_16x16x32_f16(
                       false, am0, false, bm, (short)0, acc0, false, false);
            acc1 = __builtin_amdgcn_wmma_f32_16x16x32_f16(
                       false, am1, false, bm, (short)0, acc1, false, false);
        }
    }

    // ---- epilogue: C/D layout VGPR g -> M = g + (lane<16 ? 0 : 8), N = nl
    const int mofs = (lane < 16) ? 0 : 8;
    float* outp = out + (size_t)b * Cout * Hout * Wout + (size_t)oh * Wout + ow;
    #pragma unroll
    for (int g = 0; g < 8; ++g) {
        int m0 = o0 + mofs + g;
        int m1 = o0 + 16 + mofs + g;
        outp[(size_t)m0 * Hout * Wout] = acc0[g] + bias[m0];
        outp[(size_t)m1 * Hout * Wout] = acc1[g] + bias[m1];
    }
}

extern "C" void kernel_launch(void* const* d_in, const int* in_sizes, int n_in,
                              void* d_out, int out_size, void* d_ws, size_t ws_size,
                              hipStream_t stream) {
    const float* x    = (const float*)d_in[0];   // (32,384,64,64)
    // d_in[1] time_emb, d_in[2] y: unused by reference
    const float* wgt  = (const float*)d_in[3];   // (384,384,3,3)
    const float* bias = (const float*)d_in[4];   // (384,)
    float* out = (float*)d_out;                  // (32,384,32,32)

    dim3 grid(Cout / MTILE,                       // 12  (innermost: share x in L2)
              (Batch * Hout * Wout) / NTILE);     // 256
    dim3 block(256);
    downsample_ternary_wmma<<<grid, block, 0, stream>>>(x, wgt, bias, out);
}
